// AwarePositionRegMatcher_12103217840340
// MI455X (gfx1250) — compile-verified, matrix operand order checked by reading, not run on hardware
//
#include <hip/hip_runtime.h>
#include <hip/hip_bf16.h>

#define NN 1024
#define BB 4
#define CC 256
#define HH 8
#define DD 32

typedef __attribute__((ext_vector_type(16))) __bf16 v16bf;
typedef __attribute__((ext_vector_type(8)))  float  v8f;

// ---- WMMA wrapper: D = A(16x32 bf16) * B(32x16 bf16) + C(16x16 f32) ----
__device__ __forceinline__ v8f wmma32(v16bf a, v16bf b, v8f c) {
  return __builtin_amdgcn_wmma_f32_16x16x32_bf16(
      /*neg_a=*/false, a, /*neg_b=*/false, b,
      /*c_mod=*/(short)0, c, /*reuse_a=*/false, /*reuse_b=*/false);
}

// ---- Load 16x32 16-bit A tile (or B tile from row-major [N][K] source) ----
// ISA layout: lane<16: M=lane, K = {0..7} in elems 0..7, {16..23} in elems 8..15
//             lane>=16: M=lane-16, K = {8..15}, {24..31}
template <typename T>
__device__ __forceinline__ v16bf ld16_rm(const T* __restrict__ src, int ld) {
  const int lane = threadIdx.x & 31;
  const int r = lane & 15;
  const int kb = (lane >> 4) << 3;   // 0 or 8
  const T* p = src + (size_t)r * ld + kb;
  v16bf a;
#pragma unroll
  for (int e = 0; e < 8; ++e) a[e] = (__bf16)p[e];
#pragma unroll
  for (int e = 0; e < 8; ++e) a[8 + e] = (__bf16)p[16 + e];
  return a;
}

// ---- Load 32x16 16-bit B tile from K-major [K][N] source ----
template <typename T>
__device__ __forceinline__ v16bf ld16_km(const T* __restrict__ src, int ld) {
  const int lane = threadIdx.x & 31;
  const int n = lane & 15;
  const int kb = (lane >> 4) << 3;
  const T* p = src + (size_t)kb * ld + n;
  v16bf a;
#pragma unroll
  for (int e = 0; e < 8; ++e) a[e] = (__bf16)p[(size_t)e * ld];
#pragma unroll
  for (int e = 0; e < 8; ++e) a[8 + e] = (__bf16)p[(size_t)(16 + e) * ld];
  return a;
}

// ================= Kernel 1: projections + cosine norm ==================
// grid (NB/16, 3), block 256. Wave w handles head h=w (cols [32w,32w+32)).
// Output layout [B,H,N,D] bf16.
__global__ void __launch_bounds__(256)
proj_kernel(const float* __restrict__ q, const float* __restrict__ k,
            const float* __restrict__ v, const float* __restrict__ Wq,
            const float* __restrict__ Wk, const float* __restrict__ Wv,
            __bf16* __restrict__ qo, __bf16* __restrict__ ko,
            __bf16* __restrict__ vo) {
  const int which = blockIdx.y;
  const float* X = (which == 0) ? q : (which == 1) ? k : v;
  const float* W = (which == 0) ? Wq : (which == 1) ? Wk : Wv;
  __bf16* O = (which == 0) ? qo : (which == 1) ? ko : vo;
  const bool donorm = (which < 2);

  const int w = threadIdx.x >> 5;
  const int lane = threadIdx.x & 31;
  const int hl = lane >> 4;
  const int rowbase = blockIdx.x * 16;
  const int colb = w * DD;

  v8f acc0 = {}, acc1 = {};
#pragma unroll
  for (int kk = 0; kk < CC; kk += 32) {
    v16bf a  = ld16_rm(X + (size_t)rowbase * CC + kk, CC);
    v16bf b0 = ld16_rm(W + (size_t)colb * CC + kk, CC);       // W^T tile
    v16bf b1 = ld16_rm(W + (size_t)(colb + 16) * CC + kk, CC);
    acc0 = wmma32(a, b0, acc0);
    acc1 = wmma32(a, b1, acc1);
  }
#pragma unroll
  for (int i = 0; i < 8; ++i) {
    float s = acc0[i] * acc0[i] + acc1[i] * acc1[i];
    s += __shfl_xor(s, 1); s += __shfl_xor(s, 2);
    s += __shfl_xor(s, 4); s += __shfl_xor(s, 8);   // row sum-of-sq per half
    const float rn = donorm ? rsqrtf(fmaxf(s, 1e-24f)) : 1.0f;
    const int gr = rowbase + i + hl * 8;            // flat row = n*B + b
    const int n = gr >> 2, b = gr & 3;
    const size_t base = (((size_t)b * HH + w) * NN + n) * DD;
    const int d0 = lane & 15;
    O[base + d0]      = (__bf16)(acc0[i] * rn);
    O[base + 16 + d0] = (__bf16)(acc1[i] * rn);
  }
}

// ================= Kernel 2: position bias [H,N,N] ==================
__global__ void __launch_bounds__(256)
bias_kernel(const float* __restrict__ qb, const float* __restrict__ kbx,
            const float* __restrict__ pw, const float* __restrict__ pb,
            float* __restrict__ bias) {
  __shared__ float spw[HH * 64];
  const int tid = threadIdx.x;
  spw[tid] = pw[tid];
  spw[tid + 256] = pw[tid + 256];
  __syncthreads();

  const int idx = blockIdx.x * 256 + tid;
  const int n = idx >> 10, m = idx & 1023;

  const float x1 = qb[n * 4 + 0], y1 = qb[n * 4 + 1];
  const float x2 = qb[n * 4 + 2], y2 = qb[n * 4 + 3];
  const float cx = 0.5f * (x1 + x2), cy = 0.5f * (y1 + y2);
  const float ww = x2 - x1 + 1.0f, hh = y2 - y1 + 1.0f;
  const float a1 = kbx[m * 4 + 0], b1 = kbx[m * 4 + 1];
  const float a2 = kbx[m * 4 + 2], b2 = kbx[m * 4 + 3];
  const float cxr = 0.5f * (a1 + a2), cyr = 0.5f * (b1 + b2);
  const float wr = a2 - a1 + 1.0f, hr = b2 - b1 + 1.0f;

  float pm[4];
  pm[0] = __logf(fabsf((cx - cxr) / ww) + 1e-3f);
  pm[1] = __logf(fabsf((cy - cyr) / hh) + 1e-3f);
  pm[2] = __logf(ww / wr);
  pm[3] = __logf(hh / hr);

  float pe[8];
#pragma unroll
  for (int h = 0; h < 8; ++h) pe[h] = pb[h];

#pragma unroll
  for (int j = 0; j < 8; ++j) {
    const float invd = exp2f(-1.2457230f * (float)j);  // 1000^(-j/8)
#pragma unroll
    for (int c = 0; c < 4; ++c) {
      const float f = pm[c] * 100.0f * invd;
      const float sn = __sinf(f), cs = __cosf(f);
      const int cb = c * 16 + j;
#pragma unroll
      for (int h = 0; h < 8; ++h)
        pe[h] += sn * spw[h * 64 + cb] + cs * spw[h * 64 + cb + 8];
    }
  }
#pragma unroll
  for (int h = 0; h < 8; ++h)
    bias[((size_t)h * NN + n) * NN + m] = __logf(fmaxf(pe[h], 0.0f) + 1e-6f);
}

// ================= Kernel 3: attention ==================
// grid (N/16, H, B), block 256 (8 waves). Wave w owns m-chunk [128w,128w+128).
__global__ void __launch_bounds__(256)
attn_kernel(const __bf16* __restrict__ qw, const __bf16* __restrict__ kw,
            const __bf16* __restrict__ vw, const float* __restrict__ bias,
            float* __restrict__ out, float* __restrict__ attn_out) {
  __shared__ float redA[16][8];
  __shared__ float redB[16][8];
  __shared__ __bf16 pstage[8][16][128];
  __shared__ float opart[8][16][32];

  const int b = blockIdx.z, h = blockIdx.y, nb = blockIdx.x * 16;
  const int w = threadIdx.x >> 5, lane = threadIdx.x & 31;
  const int hl = lane >> 4, nlo = lane & 15;
  const size_t bh = ((size_t)b * HH + h) * NN;

  const v16bf aq = ld16_rm(qw + (bh + nb) * DD, DD);
  v8f acc[8] = {};
  const int mbase = w * 128;
#pragma unroll
  for (int t = 0; t < 8; ++t) {
    v16bf bk = ld16_rm(kw + (bh + mbase + t * 16) * DD, DD);  // k^T tile
    acc[t] = wmma32(aq, bk, acc[t]);
  }
  // ---- softmax: row max ----
  float gm[8], inv[8];
#pragma unroll
  for (int i = 0; i < 8; ++i) {
    float rm = acc[0][i];
#pragma unroll
    for (int t = 1; t < 8; ++t) rm = fmaxf(rm, acc[t][i]);
    rm = fmaxf(rm, __shfl_xor(rm, 1)); rm = fmaxf(rm, __shfl_xor(rm, 2));
    rm = fmaxf(rm, __shfl_xor(rm, 4)); rm = fmaxf(rm, __shfl_xor(rm, 8));
    if (nlo == 0) redA[i + hl * 8][w] = rm;
  }
  __syncthreads();
#pragma unroll
  for (int i = 0; i < 8; ++i) {
    const int row = i + hl * 8;
    float m = redA[row][0];
#pragma unroll
    for (int ww = 1; ww < 8; ++ww) m = fmaxf(m, redA[row][ww]);
    gm[i] = m;
  }
  // ---- exp + row sum ----
#pragma unroll
  for (int i = 0; i < 8; ++i) {
    float s = 0.0f;
#pragma unroll
    for (int t = 0; t < 8; ++t) {
      acc[t][i] = __expf(acc[t][i] - gm[i]);
      s += acc[t][i];
    }
    s += __shfl_xor(s, 1); s += __shfl_xor(s, 2);
    s += __shfl_xor(s, 4); s += __shfl_xor(s, 8);
    if (nlo == 0) redB[i + hl * 8][w] = s;
  }
  __syncthreads();
#pragma unroll
  for (int i = 0; i < 8; ++i) {
    const int row = i + hl * 8;
    float s = 0.0f;
#pragma unroll
    for (int ww = 0; ww < 8; ++ww) s += redB[row][ww];
    inv[i] = 1.0f / s;
  }
  // ---- attn = softmax + bias : store to d_out, stage bf16 for P@v ----
#pragma unroll
  for (int t = 0; t < 8; ++t) {
#pragma unroll
    for (int i = 0; i < 8; ++i) {
      const int row = i + hl * 8;
      const int n = nb + row;
      const int m = mbase + t * 16 + nlo;
      const float av =
          acc[t][i] * inv[i] + bias[((size_t)h * NN + n) * NN + m];
      attn_out[(bh + n) * NN + m] = av;
      pstage[w][row][t * 16 + nlo] = (__bf16)av;
    }
  }
  // ---- O partial = P(16x128) @ v(128x32) ----
  v8f o0 = {}, o1 = {};
#pragma unroll
  for (int s = 0; s < 4; ++s) {
    v16bf ap = ld16_rm(&pstage[w][0][s * 32], 128);
    const __bf16* vbase = vw + (bh + mbase + s * 32) * DD;
    v16bf bv0 = ld16_km(vbase, DD);
    v16bf bv1 = ld16_km(vbase + 16, DD);
    o0 = wmma32(ap, bv0, o0);
    o1 = wmma32(ap, bv1, o1);
  }
#pragma unroll
  for (int i = 0; i < 8; ++i) {
    const int row = i + hl * 8;
    opart[w][row][nlo] = o0[i];
    opart[w][row][16 + nlo] = o1[i];
  }
  __syncthreads();
  // ---- cross-wave reduction of O, write [N,B,C] ----
#pragma unroll
  for (int e = 0; e < 2; ++e) {
    const int idx = threadIdx.x + e * 256;
    const int row = idx >> 5, d = idx & 31;
    float s = 0.0f;
#pragma unroll
    for (int ww = 0; ww < 8; ++ww) s += opart[ww][row][d];
    const int n = nb + row;
    out[((size_t)n * BB + b) * CC + h * DD + d] = s;
  }
}

extern "C" void kernel_launch(void* const* d_in, const int* in_sizes, int n_in,
                              void* d_out, int out_size, void* d_ws,
                              size_t ws_size, hipStream_t stream) {
  const float* query  = (const float*)d_in[0];
  const float* key    = (const float*)d_in[1];
  const float* value  = (const float*)d_in[2];
  const float* qboxes = (const float*)d_in[3];
  const float* kboxes = (const float*)d_in[4];
  const float* Wq     = (const float*)d_in[5];
  const float* Wk     = (const float*)d_in[6];
  const float* Wv     = (const float*)d_in[7];
  const float* posw   = (const float*)d_in[8];
  const float* posb   = (const float*)d_in[9];

  char* ws = (char*)d_ws;
  __bf16* qw = (__bf16*)(ws);                          // 2 MB
  __bf16* kw = (__bf16*)(ws + (2ull << 20));           // 2 MB
  __bf16* vw = (__bf16*)(ws + (4ull << 20));           // 2 MB
  float*  bias = (float*)(ws + (6ull << 20));          // 32 MB [H,N,N]

  float* out = (float*)d_out;                          // [N,B,C]
  float* attn = out + (size_t)NN * BB * CC;            // [B,H,N,N]

  proj_kernel<<<dim3((NN * BB) / 16, 3), 256, 0, stream>>>(
      query, key, value, Wq, Wk, Wv, qw, kw, vw);
  bias_kernel<<<dim3((NN * NN) / 256), 256, 0, stream>>>(
      qboxes, kboxes, posw, posb, bias);
  attn_kernel<<<dim3(NN / 16, HH, BB), 256, 0, stream>>>(
      qw, kw, vw, bias, out, attn);
}
